// GCN3D_Feb12_25520695673033
// MI455X (gfx1250) — matrix operand order repaired and checked by reading, not compiled, not run on hardware
//
#include <hip/hip_runtime.h>
#include <hip/hip_bf16.h>

// ---------------------------------------------------------------------------
// CDNA5 (gfx1250) WMMA types
// ---------------------------------------------------------------------------
typedef __attribute__((ext_vector_type(16))) __bf16 v16bf;
typedef __attribute__((ext_vector_type(8)))  float  v8f;

__device__ __forceinline__ float eluf(float x) {
    return x > 0.f ? x : (__expf(x) - 1.f);
}

#define WMMA_BF16(a, b, c) \
    __builtin_amdgcn_wmma_f32_16x16x32_bf16(false, (a), false, (b), (short)0, (c), false, false)

// ---------------------------------------------------------------------------
// Pack fp32 W[K,Nc] (row major) into bf16 B-fragment panels:
//   Wp[(kt*Nc + col)*32 + kk] = W[(kt*32+kk)*Nc + col]   (zero-padded past K)
// A lane's 16-element B fragment is then 32 contiguous bytes at
//   ((k0/32)*Nc + col)*32 + half*16
// in the contiguous K order this file uses for both A and B (lanes 0-15: K k0..k0+15, lanes 16-31:
// K k0+16..k0+31).
// ---------------------------------------------------------------------------
__global__ void k_packW(const float* __restrict__ W, __bf16* __restrict__ Wp,
                        int K, int Nc, int Kt)
{
    long t = (long)blockIdx.x * blockDim.x + threadIdx.x;
    const long total = (long)Kt * Nc * 32;
    if (t >= total) return;
    const int  kk   = (int)(t & 31);
    const long rest = t >> 5;
    const int  col  = (int)(rest % Nc);
    const int  kt   = (int)(rest / Nc);
    const int  k    = kt * 32 + kk;
    Wp[t] = (k < K) ? (__bf16)W[(long)k * Nc + col] : (__bf16)0.f;
}

// ---------------------------------------------------------------------------
// bf16-operand / f32-accum WMMA GEMM:  C[M,Nc] (+)= op(A[M,Kp] @ W + bias)
// Kp multiple of 32; lda = Kp. Wp = packed panels. flags: 1=bias, 2=ELU,
// 4=accumulate. gidx: optional row gather for A. Wave -> 16x16 tile.
// ---------------------------------------------------------------------------
__global__ void k_gemm_bf16(const float* __restrict__ A, const int* __restrict__ gidx,
                            const __bf16* __restrict__ Wp, const float* __restrict__ bias,
                            float* __restrict__ C,
                            int M, int Nc, int Kp, int flags)
{
    const int lane = threadIdx.x & 31;
    const int wave = threadIdx.x >> 5;
    const int ntN  = Nc >> 4;
    const int ntM  = M >> 4;
    const int tile = blockIdx.x * (blockDim.x >> 5) + wave;
    if (tile >= ntM * ntN) return;                 // wave-uniform (EXEC all-1s inside)
    const int tM = tile / ntN;
    const int tN = tile % ntN;

    const int m    = lane & 15;
    const int half = lane >> 4;
    const int row  = tM * 16 + m;
    const int arow = gidx ? gidx[row] : row;
    const int col  = tN * 16 + m;

    const float*  arowp = A + (long)arow * Kp + half * 16;  // contiguous K base for this lane
    const __bf16* wcol  = Wp + (long)col * 32 + half * 16;
    const long    wstep = (long)Nc * 32;

    v8f acc = {};
    for (int k0 = 0; k0 < Kp; k0 += 32) {
        const float4 a01 = *(const float4*)(arowp + k0);
        const float4 a23 = *(const float4*)(arowp + k0 + 4);
        const float4 a45 = *(const float4*)(arowp + k0 + 8);
        const float4 a67 = *(const float4*)(arowp + k0 + 8 + 4);
        const v16bf b = *(const v16bf*)(wcol + (long)(k0 >> 5) * wstep);

        v16bf a;
        a[0]  = (__bf16)a01.x;  a[1]  = (__bf16)a01.y;
        a[2]  = (__bf16)a01.z;  a[3]  = (__bf16)a01.w;
        a[4]  = (__bf16)a23.x;  a[5]  = (__bf16)a23.y;
        a[6]  = (__bf16)a23.z;  a[7]  = (__bf16)a23.w;
        a[8]  = (__bf16)a45.x;  a[9]  = (__bf16)a45.y;
        a[10] = (__bf16)a45.z;  a[11] = (__bf16)a45.w;
        a[12] = (__bf16)a67.x;  a[13] = (__bf16)a67.y;
        a[14] = (__bf16)a67.z;  a[15] = (__bf16)a67.w;

        acc = WMMA_BF16(a, b, acc);
    }

#pragma unroll
    for (int r = 0; r < 8; ++r) {                  // C/D: VGPR r -> M=half*8+r, N=lane&15
        const int  orow = tM * 16 + half * 8 + r;
        const long off  = (long)orow * Nc + col;
        float v = acc[r];
        if (flags & 4) v += C[off];
        if (flags & 1) v += bias[col];
        if (flags & 2) v = eluf(v);
        C[off] = v;
    }
}

// ---------------------------------------------------------------------------
// Memory-bound helpers
// ---------------------------------------------------------------------------
__global__ void k_fill(float* __restrict__ p, float v, long n) {
    long t = (long)blockIdx.x * blockDim.x + threadIdx.x;
    if (t < n) p[t] = v;
}

__global__ void k_padx(const float* __restrict__ x, float* __restrict__ xpad, int N) {
    long t = (long)blockIdx.x * blockDim.x + threadIdx.x;
    if (t >= (long)N * 16) return;
    const long i = t >> 4;
    const int  c = (int)(t & 15);
    xpad[i * 32 + c] = x[t];
}

__global__ void k_edge_deg(float* __restrict__ deg, const int* __restrict__ dst, int E) {
    long t = (long)blockIdx.x * blockDim.x + threadIdx.x;
    if (t < E) atomicAdd(&deg[dst[t]], 1.0f);
}

__global__ void k_rsqrt_inplace(float* __restrict__ d, long n) {
    long t = (long)blockIdx.x * blockDim.x + threadIdx.x;
    if (t < n) d[t] = rsqrtf(d[t]);
}

__global__ void k_edge_scatter(const float* __restrict__ h, const int* __restrict__ src,
                               const int* __restrict__ dst, const float* __restrict__ dinv,
                               float* __restrict__ out, int E, int C)
{
    long t = (long)blockIdx.x * blockDim.x + threadIdx.x;
    const int q = C >> 2;
    if (t >= (long)E * q) return;
    const int e = (int)(t / q);
    const int c = (int)(t % q) << 2;
    const int s = src[e], d = dst[e];
    const float coef = dinv[s] * dinv[d];
    const float4 hv = *(const float4*)(h + (long)s * C + c);
    float* op = out + (long)d * C + c;
    atomicAdd(op + 0, hv.x * coef);
    atomicAdd(op + 1, hv.y * coef);
    atomicAdd(op + 2, hv.z * coef);
    atomicAdd(op + 3, hv.w * coef);
}

__global__ void k_selfloop(const float* __restrict__ agg, const float* __restrict__ h,
                           const float* __restrict__ dinv, const float* __restrict__ bias,
                           float* __restrict__ out, long NC, int C)
{
    long t = (long)blockIdx.x * blockDim.x + threadIdx.x;
    if (t >= NC) return;
    const long i = t / C;
    const int  c = (int)(t % C);
    const float di = dinv[i];
    out[t] = eluf(agg[t] + h[t] * di * di + bias[c]);
}

__global__ void k_in_stats(const float* __restrict__ x, float* __restrict__ mean,
                           float* __restrict__ rstd, int NPG, int C)
{
    __shared__ float s1[256], s2[256];
    const int g = blockIdx.x / C, c = blockIdx.x % C;
    const float* p = x + (long)g * NPG * C + c;
    float sum = 0.f, sq = 0.f;
    for (int i = threadIdx.x; i < NPG; i += blockDim.x) {
        const float v = p[(long)i * C];
        sum += v; sq += v * v;
    }
    s1[threadIdx.x] = sum; s2[threadIdx.x] = sq;
    __syncthreads();
    for (int s = blockDim.x >> 1; s > 0; s >>= 1) {
        if ((int)threadIdx.x < s) {
            s1[threadIdx.x] += s1[threadIdx.x + s];
            s2[threadIdx.x] += s2[threadIdx.x + s];
        }
        __syncthreads();
    }
    if (threadIdx.x == 0) {
        const float mu  = s1[0] / (float)NPG;
        const float var = s2[0] / (float)NPG - mu * mu;
        mean[blockIdx.x] = mu;
        rstd[blockIdx.x] = rsqrtf(var + 1e-5f);
    }
}

__global__ void k_in_norm(const float* __restrict__ x, const float* __restrict__ mean,
                          const float* __restrict__ rstd, float* __restrict__ y,
                          long NC, int NPG, int C)
{
    long t = (long)blockIdx.x * blockDim.x + threadIdx.x;
    if (t >= NC) return;
    const long i = t / C;
    const int  c = (int)(t % C);
    const int  g = (int)(i / NPG);
    y[t] = (x[t] - mean[g * C + c]) * rstd[g * C + c];
}

__global__ void k_bc(const int* __restrict__ cluster, const int* __restrict__ batch,
                     int* __restrict__ bcA, float* __restrict__ cnt, int N)
{
    long t = (long)blockIdx.x * blockDim.x + threadIdx.x;
    if (t >= N) return;
    const int bc = cluster[t] + batch[t] * 512;
    bcA[t] = bc;
    atomicAdd(&cnt[bc], 1.0f);
}

__global__ void k_pool_sum(const float* __restrict__ y, const int* __restrict__ bcA,
                           float* __restrict__ pooled, int N, int C)
{
    long t = (long)blockIdx.x * blockDim.x + threadIdx.x;
    const int q = C >> 2;
    if (t >= (long)N * q) return;
    const int i = (int)(t / q);
    const int c = (int)(t % q) << 2;
    const int bc = bcA[i];
    const float4 yv = *(const float4*)(y + (long)i * C + c);
    float* pp = pooled + (long)bc * C + c;
    atomicAdd(pp + 0, yv.x);
    atomicAdd(pp + 1, yv.y);
    atomicAdd(pp + 2, yv.z);
    atomicAdd(pp + 3, yv.w);
}

__global__ void k_pool_div(float* __restrict__ pooled, const float* __restrict__ cnt,
                           int CT, int C)
{
    long t = (long)blockIdx.x * blockDim.x + threadIdx.x;
    if (t >= (long)CT * C) return;
    pooled[t] /= fmaxf(cnt[t / C], 1.0f);
}

__global__ void k_adj(float* __restrict__ A, const int* __restrict__ bcA,
                      const int* __restrict__ src, const int* __restrict__ dst,
                      int E, int CT)
{
    long t = (long)blockIdx.x * blockDim.x + threadIdx.x;
    if (t >= E) return;
    A[(long)bcA[src[t]] * CT + bcA[dst[t]]] = 1.0f;
}

__global__ void k_diag0(float* __restrict__ A, int CT) {
    long t = (long)blockIdx.x * blockDim.x + threadIdx.x;
    if (t < CT) A[t * CT + t] = 0.0f;
}

__global__ void k_cdeg(const float* __restrict__ A, float* __restrict__ cdinv, int CT)
{
    __shared__ float s[256];
    const int j = blockIdx.x;
    float sum = 0.f;
    for (int i = threadIdx.x; i < CT; i += blockDim.x) sum += A[(long)i * CT + j];
    s[threadIdx.x] = sum;
    __syncthreads();
    for (int st = blockDim.x >> 1; st > 0; st >>= 1) {
        if ((int)threadIdx.x < st) s[threadIdx.x] += s[threadIdx.x + st];
        __syncthreads();
    }
    if (threadIdx.x == 0) cdinv[j] = rsqrtf(sum + 1.0f);
}

// AnT[j,i] = A[i,j] * cdinv[i] * cdinv[j]
__global__ void k_ant(const float* __restrict__ A, const float* __restrict__ cdinv,
                      float* __restrict__ AnT, int CT)
{
    long t = (long)blockIdx.x * blockDim.x + threadIdx.x;
    if (t >= (long)CT * CT) return;
    const int j = (int)(t / CT);
    const int i = (int)(t % CT);
    AnT[t] = A[(long)i * CT + j] * cdinv[i] * cdinv[j];
}

__global__ void k_final(const float* __restrict__ z, const float* __restrict__ Wf,
                        const float* __restrict__ bf, float* __restrict__ out, int N)
{
    long t = (long)blockIdx.x * blockDim.x + threadIdx.x;
    if (t >= (long)N * 3) return;
    const int i = (int)(t / 3), o = (int)(t % 3);
    float s = bf[o];
    const float* zp = z + (long)i * 128;
#pragma unroll 8
    for (int k = 0; k < 128; ++k) s += zp[k] * Wf[k * 3 + o];
    out[t] = s;
}

// ---------------------------------------------------------------------------
static inline unsigned cdiv(long a, long b) { return (unsigned)((a + b - 1) / b); }

extern "C" void kernel_launch(void* const* d_in, const int* in_sizes, int n_in,
                              void* d_out, int out_size, void* d_ws, size_t ws_size,
                              hipStream_t stream)
{
    (void)n_in; (void)out_size; (void)ws_size;
    const float* x        = (const float*)d_in[0];
    const int*   adj      = (const int*)d_in[1];
    const int*   in_batch = (const int*)d_in[3];
    const int*   cluster  = (const int*)d_in[4];
    const float* Wg1  = (const float*)d_in[5];   const float* bg1  = (const float*)d_in[6];
    const float* Wf1  = (const float*)d_in[7];   const float* bf1  = (const float*)d_in[8];
    const float* Wl1  = (const float*)d_in[9];   const float* bl1  = (const float*)d_in[10];
    const float* Wfl1 = (const float*)d_in[11];  const float* bfl1 = (const float*)d_in[12];
    const float* Wo1  = (const float*)d_in[13];  const float* bo1  = (const float*)d_in[14];
    const float* Wfo1 = (const float*)d_in[15];  const float* bfo1 = (const float*)d_in[16];
    float* out = (float*)d_out;

    const int N   = in_sizes[0] / 16;       // 100000
    const int E   = in_sizes[1] / 2;        // 800000
    const int G   = 4;
    const int NPG = N / G;                  // 25000
    const int CT  = G * 512;                // 2048
    const int* src = adj;
    const int* dst = adj + E;

    // ---- workspace carve-up ----
    float* buf0   = (float*)d_ws;                   // [N,256]
    float* buf1   = buf0 + (long)N * 256;           // [N,256]
    float* xpad   = buf1 + (long)N * 256;           // [N,32]
    float* dinv   = xpad + (long)N * 32;            // [N]
    int*   bcA    = (int*)(dinv + N);               // [N]
    float* pooled = (float*)(bcA + N);              // [CT,256]
    float* cnt    = pooled + (long)CT * 256;        // [CT]
    float* Amat   = cnt + CT;                       // [CT,CT]
    float* AnT    = Amat + (long)CT * CT;           // [CT,CT]
    float* hp     = AnT + (long)CT * CT;            // [CT,128]
    float* lx     = hp + (long)CT * 128;            // [CT,128]
    float* lx2    = lx + (long)CT * 128;            // [CT,128]
    float* cdinv  = lx2 + (long)CT * 128;           // [CT]
    float* meanA  = cdinv + CT;                     // [G*256]
    float* rstdA  = meanA + G * 256;                // [G*256]
    __bf16* Wg1p  = (__bf16*)(rstdA + G * 256);     // 1*256*32
    __bf16* Wf1p  = Wg1p  + 1 * 256 * 32;           // 8*256*32
    __bf16* Wl1p  = Wf1p  + 8 * 256 * 32;           // 8*128*32
    __bf16* Wfl1p = Wl1p  + 8 * 128 * 32;           // 4*128*32
    __bf16* WoTp  = Wfl1p + 4 * 128 * 32;           // 8*128*32
    __bf16* WoBp  = WoTp  + 8 * 128 * 32;           // 4*128*32
    __bf16* hpPan = WoBp  + 4 * 128 * 32;           // 64*128*32  packed hp panels

    const int T = 256;

    // 0) pack static weights into bf16 B panels; zero-pad x
    k_packW<<<cdiv(1L * 256 * 32, T), T, 0, stream>>>(Wg1,             Wg1p,  16,  256, 1);
    k_packW<<<cdiv(8L * 256 * 32, T), T, 0, stream>>>(Wf1,             Wf1p,  256, 256, 8);
    k_packW<<<cdiv(8L * 128 * 32, T), T, 0, stream>>>(Wl1,             Wl1p,  256, 128, 8);
    k_packW<<<cdiv(4L * 128 * 32, T), T, 0, stream>>>(Wfl1,            Wfl1p, 128, 128, 4);
    k_packW<<<cdiv(8L * 128 * 32, T), T, 0, stream>>>(Wo1,             WoTp,  256, 128, 8);
    k_packW<<<cdiv(4L * 128 * 32, T), T, 0, stream>>>(Wo1 + 256 * 128, WoBp,  128, 128, 4);
    k_fill<<<cdiv((long)N * 32, T), T, 0, stream>>>(xpad, 0.0f, (long)N * 32);
    k_padx<<<cdiv((long)N * 16, T), T, 0, stream>>>(x, xpad, N);

    // 1) node degrees -> dinv
    k_fill<<<cdiv(N, T), T, 0, stream>>>(dinv, 1.0f, N);
    k_edge_deg<<<cdiv(E, T), T, 0, stream>>>(dinv, dst, E);
    k_rsqrt_inplace<<<cdiv(N, T), T, 0, stream>>>(dinv, N);

    // 2) hW = x @ Wg1 (Kp=32)
    k_gemm_bf16<<<cdiv((long)(N / 16) * 16, 8), T, 0, stream>>>(
        xpad, nullptr, Wg1p, nullptr, buf0, N, 256, 32, 0);

    // 3) GCN edge aggregation + self-loop + bias + ELU -> h (buf1)
    k_fill<<<cdiv((long)N * 256, T), T, 0, stream>>>(buf1, 0.0f, (long)N * 256);
    k_edge_scatter<<<cdiv((long)E * 64, T), T, 0, stream>>>(buf0, src, dst, dinv, buf1, E, 256);
    k_selfloop<<<cdiv((long)N * 256, T), T, 0, stream>>>(buf1, buf0, dinv, bg1, buf1,
                                                         (long)N * 256, 256);

    // 4) h2 = elu(h @ Wf1 + bf1) -> buf0
    k_gemm_bf16<<<cdiv((long)(N / 16) * 16, 8), T, 0, stream>>>(
        buf1, nullptr, Wf1p, bf1, buf0, N, 256, 256, 3);

    // 5) instance norm -> y (buf1)
    k_in_stats<<<G * 256, T, 0, stream>>>(buf0, meanA, rstdA, NPG, 256);
    k_in_norm<<<cdiv((long)N * 256, T), T, 0, stream>>>(buf0, meanA, rstdA, buf1,
                                                        (long)N * 256, NPG, 256);

    // 6) cluster mean-pool -> pooled [CT,256]
    k_fill<<<cdiv(CT, T), T, 0, stream>>>(cnt, 0.0f, CT);
    k_fill<<<cdiv((long)CT * 256, T), T, 0, stream>>>(pooled, 0.0f, (long)CT * 256);
    k_bc<<<cdiv(N, T), T, 0, stream>>>(cluster, in_batch, bcA, cnt, N);
    k_pool_sum<<<cdiv((long)N * 64, T), T, 0, stream>>>(buf1, bcA, pooled, N, 256);
    k_pool_div<<<cdiv((long)CT * 256, T), T, 0, stream>>>(pooled, cnt, CT, 256);

    // 7) coarse adjacency, degree, normalized transpose
    k_fill<<<cdiv((long)CT * CT, T), T, 0, stream>>>(Amat, 0.0f, (long)CT * CT);
    k_adj<<<cdiv(E, T), T, 0, stream>>>(Amat, bcA, src, dst, E, CT);
    k_diag0<<<cdiv(CT, T), T, 0, stream>>>(Amat, CT);
    k_cdeg<<<CT, T, 0, stream>>>(Amat, cdinv, CT);
    k_ant<<<cdiv((long)CT * CT, T), T, 0, stream>>>(Amat, cdinv, AnT, CT);

    // 8) coarse GCN via WMMA: hp = pooled @ Wl1; pack hp; lx = AnT @ hp; +self
    k_gemm_bf16<<<cdiv((long)(CT / 16) * 8, 8), T, 0, stream>>>(
        pooled, nullptr, Wl1p, nullptr, hp, CT, 128, 256, 0);
    k_packW<<<cdiv(64L * 128 * 32, T), T, 0, stream>>>(hp, hpPan, 2048, 128, 64);
    k_gemm_bf16<<<cdiv((long)(CT / 16) * 8, 8), T, 0, stream>>>(
        AnT, nullptr, hpPan, nullptr, lx, CT, 128, 2048, 0);
    k_selfloop<<<cdiv((long)CT * 128, T), T, 0, stream>>>(lx, hp, cdinv, bl1, lx,
                                                          (long)CT * 128, 128);

    // 9) lx2 = elu(lx @ Wfl1 + bfl1)
    k_gemm_bf16<<<cdiv((long)(CT / 16) * 8, 8), T, 0, stream>>>(
        lx, nullptr, Wfl1p, bfl1, lx2, CT, 128, 128, 3);

    // 10) zW = y @ Wo1[0:256,:] + gather(lx2, bc) @ Wo1[256:384,:]
    float* zW   = buf0;
    float* zagg = buf0 + (long)N * 128;
    k_gemm_bf16<<<cdiv((long)(N / 16) * 8, 8), T, 0, stream>>>(
        buf1, nullptr, WoTp, nullptr, zW, N, 128, 256, 0);
    k_gemm_bf16<<<cdiv((long)(N / 16) * 8, 8), T, 0, stream>>>(
        lx2, bcA, WoBp, nullptr, zW, N, 128, 128, 4);

    // 11) final GCN edge conv (128 ch) + ELU
    k_fill<<<cdiv((long)N * 128, T), T, 0, stream>>>(zagg, 0.0f, (long)N * 128);
    k_edge_scatter<<<cdiv((long)E * 32, T), T, 0, stream>>>(zW, src, dst, dinv, zagg, E, 128);
    k_selfloop<<<cdiv((long)N * 128, T), T, 0, stream>>>(zagg, zW, dinv, bo1, zagg,
                                                         (long)N * 128, 128);

    // 12) out = z @ Wfo1 + bfo1
    k_final<<<cdiv((long)N * 3, T), T, 0, stream>>>(zagg, Wfo1, bfo1, out, N);
}
